// Raim_49143015801324
// MI455X (gfx1250) — matrix operand order repaired and checked by reading, not compiled, not run on hardware
//
#include <hip/hip_runtime.h>

// ---------------------------------------------------------------------------
// RAIM recurrent attention-GRU for MI455X (gfx1250, wave32, WMMA bf16)
// ---------------------------------------------------------------------------
constexpr int NB = 256, NT = 48, NC = 20, ND = 256, NDEMO = 76;
constexpr int BM = 16;        // batch rows per workgroup (one WMMA M tile)
constexpr int NTHREADS = 512; // 16 waves

typedef __attribute__((ext_vector_type(16))) __bf16          v16bf;
typedef __attribute__((ext_vector_type(8)))  float           v8f;
typedef __attribute__((ext_vector_type(8)))  unsigned short  ushort8;
typedef __attribute__((ext_vector_type(16))) unsigned short  ushort16;

__device__ __forceinline__ unsigned short f2bf(float f) {
  unsigned int u = __float_as_uint(f);
  u += 0x7FFFu + ((u >> 16) & 1u);           // round-to-nearest-even
  return (unsigned short)(u >> 16);
}

// A-matrix fragment (16x32 bf16) from LDS, row-major [16][ND] halves.
// Lanes hold K = sel*16 + 0..15 (contiguous, matches repacked B).
__device__ __forceinline__ v16bf load_a_frag(const unsigned short* base, int lane, int kt) {
  int m = lane & 15, sel = lane >> 4;
  const unsigned short* p = base + m * ND + kt * 32 + sel * 16;
  ushort8 lo = *(const ushort8*)p;           // ds_load_b128
  ushort8 hi = *(const ushort8*)(p + 8);     // ds_load_b128
  ushort16 u;
#pragma unroll
  for (int j = 0; j < 8; ++j) { u[j] = lo[j]; u[8 + j] = hi[j]; }
  return __builtin_bit_cast(v16bf, u);
}

// B-matrix fragment: pre-packed so each lane's 16 halves are contiguous (32B).
__device__ __forceinline__ v16bf load_b_frag(const unsigned short* p, int lane) {
  ushort16 u = *((const ushort16*)p + lane); // 2x global_load_b128
  return __builtin_bit_cast(v16bf, u);
}

// ---------------------------------------------------------------------------
// Prep kernel 1: embedding sums -> x[B,T,D], xw[B,T]=x.wx, x2c[B,T]
// ---------------------------------------------------------------------------
__global__ __launch_bounds__(256) void embed_kernel(
    const int* __restrict__ diag, const int* __restrict__ drug,
    const int* __restrict__ lab,  const int* __restrict__ proc,
    const float* __restrict__ emb1, const float* __restrict__ emb2,
    const float* __restrict__ emb3, const float* __restrict__ emb4,
    const float* __restrict__ wx,
    float* __restrict__ x, float* __restrict__ xw, float* __restrict__ x2c)
{
  int bt = blockIdx.x;          // b*NT + t
  int d  = threadIdx.x;         // 0..255
  const int* pd = diag + bt * NC;
  const int* pr = drug + bt * NC;
  const int* pl = lab  + bt * NC;
  const int* pp = proc + bt * NC;
  float acc = 0.f, lp = 0.f;
#pragma unroll 4
  for (int c = 0; c < NC; ++c) acc += emb1[pd[c] * ND + d];
#pragma unroll 4
  for (int c = 0; c < NC; ++c) acc += emb2[pr[c] * ND + d];
#pragma unroll 4
  for (int c = 0; c < NC; ++c) { float v = emb3[pl[c] * ND + d]; acc += v; lp += v; }
#pragma unroll 4
  for (int c = 0; c < NC; ++c) { float v = emb4[pp[c] * ND + d]; acc += v; lp += v; }
  x[bt * ND + d] = acc;
  __shared__ float red[256];
  red[d] = acc * wx[d];
  __syncthreads();
  for (int s = 128; s > 0; s >>= 1) {
    if (d < s) red[d] += red[d + s];
    __syncthreads();
  }
  if (d == 0) { xw[bt] = red[0]; x2c[bt] = lp; }   // lane d==0 holds channel 0
}

// Exclusive prefix sum of x2c along T
__global__ void csum_kernel(const float* __restrict__ x2c, float* __restrict__ csum) {
  int b = blockIdx.x * blockDim.x + threadIdx.x;
  if (b >= NB) return;
  float s = 0.f;
  for (int t = 0; t < NT; ++t) { csum[b * NT + t] = s; s += x2c[b * NT + t]; }
}

// demo_e = demo @ demo_W^T + demo_b
__global__ void demo_kernel(const float* __restrict__ demo, const float* __restrict__ demo_W,
                            const float* __restrict__ demo_b, float* __restrict__ demo_e) {
  int idx = blockIdx.x * blockDim.x + threadIdx.x;
  if (idx >= NB * ND) return;
  int b = idx >> 8, d = idx & 255;
  float s = demo_b[d];
#pragma unroll 4
  for (int j = 0; j < NDEMO; ++j) s += demo[b * NDEMO + j] * demo_W[d * NDEMO + j];
  demo_e[idx] = s;
}

// Repack row-major W[N][K] f32 into bf16 WMMA-B fragments:
// flat = ((nt*KT + kt)*32 + lane)*16 + j ; element (n = nt*16 + lane%16,
// k = kt*32 + (lane/16)*16 + j)  (contiguous-half k order, same as load_a_frag)
__global__ void repack_kernel(const float* __restrict__ W, unsigned short* __restrict__ out,
                              int N, int K) {
  int idx = blockIdx.x * blockDim.x + threadIdx.x;
  if (idx >= N * K) return;
  int j    = idx & 15;
  int lane = (idx >> 4) & 31;
  int rest = idx >> 9;
  int KT = K >> 5;
  int kt = rest % KT;
  int nt = rest / KT;
  int n = nt * 16 + (lane & 15);
  int k = kt * 32 + (lane >> 4) * 16 + j;
  out[idx] = f2bf(W[n * K + k]);
}

// ---------------------------------------------------------------------------
// Main fused recurrent kernel: one workgroup (16 waves) owns 16 batch rows
// for all T steps; hidden state lives in LDS.
// ---------------------------------------------------------------------------
__global__ __launch_bounds__(NTHREADS) void raim_main(
    const float* __restrict__ x, const float* __restrict__ xw,
    const float* __restrict__ csum, const float* __restrict__ demo_e,
    const unsigned short* __restrict__ pWih, const unsigned short* __restrict__ pWhh,
    const unsigned short* __restrict__ pWh,
    const float* __restrict__ bh, const float* __restrict__ b_ih, const float* __restrict__ b_hh,
    const float* __restrict__ out_W, const float* __restrict__ out_b,
    float* __restrict__ out)
{
  __shared__ float          h_f[BM * ND];           // f32 master hidden state
  __shared__ unsigned short h_b[BM * ND];           // bf16 copy (WMMA A)
  __shared__ unsigned short ctx_b[BM * ND];         // bf16 context (WMMA A)
  __shared__ float          gi_s[BM * 3 * ND];      // ctx @ W_ih^T
  __shared__ float          gh_s[BM * 3 * ND];      // h   @ W_hh^T
  __shared__ float          t1s[BM * NT];
  __shared__ float          as_[BM * NT];

  const int tid  = threadIdx.x;
  const int lane = tid & 31;
  const int wave = tid >> 5;
  const int b0   = blockIdx.x * BM;

  // hoist per-thread GRU biases for phase 5 (d = tid&255, constant over T)
  const int d5 = tid & 255;
  const float bir = b_ih[d5], biz = b_ih[ND + d5], bin = b_ih[2 * ND + d5];
  const float bhr = b_hh[d5], bhz = b_hh[ND + d5], bhn = b_hh[2 * ND + d5];

  for (int p = tid; p < BM * ND; p += NTHREADS) { h_f[p] = 0.f; h_b[p] = 0; }
  __syncthreads();

  for (int i = 0; i < NT; ++i) {
    // ---- phase 1: t1 = h @ Wh^T  (3 N-tiles, waves 0..2, WMMA bf16, pipelined B)
    if (wave < 3) {
      const unsigned short* Wbase = pWh + wave * 8 * 512;
      v16bf bnext = load_b_frag(Wbase, lane);
      v8f acc = {0.f,0.f,0.f,0.f,0.f,0.f,0.f,0.f};
#pragma unroll
      for (int kt = 0; kt < 8; ++kt) {
        v16bf af = load_a_frag(h_b, lane, kt);
        v16bf bcur = bnext;
        if (kt < 7) bnext = load_b_frag(Wbase + (kt + 1) * 512, lane);
        acc = __builtin_amdgcn_wmma_f32_16x16x32_bf16(false, af, false, bcur,
                                                      (short)0, acc, false, false);
      }
      const int m0 = (lane < 16) ? 0 : 8;
      const int n  = wave * 16 + (lane & 15);
#pragma unroll
      for (int r = 0; r < 8; ++r) t1s[(m0 + r) * NT + n] = acc[r];
    }
    __syncthreads();

    // ---- phase 2: add bh + g*xw, softmax over all T
    // 16 threads per row, 3 logits each, shfl reductions (width 16)
    if (tid < 256) {
      const int m = tid >> 4, sub = tid & 15, b = b0 + m;
      const bool zr = (csum[b * NT + i] == 0.f);
      float v[3];
      float mx = -3.4e38f;
#pragma unroll
      for (int k = 0; k < 3; ++k) {
        const int t = sub * 3 + k;
        float g = (t == i) ? 1.f : ((t < i && zr) ? 1.f : 0.f);
        float val = t1s[m * NT + t] + bh[t] + g * xw[b * NT + t];
        v[k] = val;
        mx = fmaxf(mx, val);
      }
#pragma unroll
      for (int off = 8; off; off >>= 1) mx = fmaxf(mx, __shfl_xor(mx, off, 16));
      float s = 0.f;
#pragma unroll
      for (int k = 0; k < 3; ++k) { v[k] = __expf(v[k] - mx); s += v[k]; }
#pragma unroll
      for (int off = 8; off; off >>= 1) s += __shfl_xor(s, off, 16);
      const float inv = 1.f / s;
#pragma unroll
      for (int k = 0; k < 3; ++k) as_[m * NT + sub * 3 + k] = v[k] * inv;
    }
    __syncthreads();

    // ---- phase 3: ctx = sum_{t<=i} a[t] * x[b,t,:]  (one wave per row)
    {
      const int m = wave;
      const float* xr = x + (b0 + m) * NT * ND;
#pragma unroll
      for (int c = 0; c < 8; ++c) {
        const int d = lane + 32 * c;
        float s = 0.f;
        for (int t = 0; t <= i; ++t) s += as_[m * NT + t] * xr[t * ND + d];
        ctx_b[m * ND + d] = f2bf(s);
      }
    }
    __syncthreads();

    // ---- phase 4: gi = ctx@W_ih^T (waves 0..7), gh = h@W_hh^T (waves 8..15)
    // 6 N-tiles per wave; B-fragment stream software-pipelined (bnext).
    {
      const bool isGi = (wave < 8);
      const unsigned short* Wp = isGi ? pWih : pWhh;
      const unsigned short* Ap = isGi ? ctx_b : h_b;
      float* G = isGi ? gi_s : gh_s;
      const int ntBase = (wave & 7) * 6;
      v16bf af[8];
#pragma unroll
      for (int kt = 0; kt < 8; ++kt) af[kt] = load_a_frag(Ap, lane, kt);
      const unsigned short* Wbase = Wp + (size_t)ntBase * 8 * 512;
      v16bf bnext = load_b_frag(Wbase, lane);
      const int m0 = (lane < 16) ? 0 : 8;
      const int nl = lane & 15;
#pragma unroll
      for (int q = 0; q < 6; ++q) {
        v8f acc = {0.f,0.f,0.f,0.f,0.f,0.f,0.f,0.f};
#pragma unroll
        for (int kt = 0; kt < 8; ++kt) {
          v16bf bcur = bnext;
          const int fi = q * 8 + kt + 1;          // next fragment in this wave's stream
          if (fi < 48) bnext = load_b_frag(Wbase + (size_t)fi * 512, lane);
          acc = __builtin_amdgcn_wmma_f32_16x16x32_bf16(false, af[kt], false, bcur,
                                                        (short)0, acc, false, false);
        }
        const int n = (ntBase + q) * 16 + nl;
#pragma unroll
        for (int r = 0; r < 8; ++r) G[(m0 + r) * (3 * ND) + n] = acc[r];
      }
    }
    __syncthreads();

    // ---- phase 5: fused GRU update (512 threads: d = tid&255, 8 rows each)
    {
      const int d = d5;
      const int mb = (tid >> 8) * 8;
#pragma unroll
      for (int mm = 0; mm < 8; ++mm) {
        const int m = mb + mm;
        const int gb = m * 3 * ND;
        float xr_ = gi_s[gb + d] + bir + gh_s[gb + d] + bhr;
        float xz_ = gi_s[gb + ND + d] + biz + gh_s[gb + ND + d] + bhz;
        float r = 1.f / (1.f + __expf(-xr_));
        float z = 1.f / (1.f + __expf(-xz_));
        float nn = tanhf(gi_s[gb + 2 * ND + d] + bin + r * (gh_s[gb + 2 * ND + d] + bhn));
        float hp = h_f[m * ND + d];
        float hv = (1.f - z) * nn + z * hp;
        h_f[m * ND + d] = hv;
        h_b[m * ND + d] = f2bf(hv);
      }
    }
    __syncthreads();
  }

  // ---- output head: (hT + demo_e) @ out_W^T + out_b
  if (tid < 32) {
    const int m = tid >> 1, cc = tid & 1, b = b0 + m;
    float s = out_b[cc];
    for (int d = 0; d < ND; ++d)
      s += (h_f[m * ND + d] + demo_e[b * ND + d]) * out_W[cc * ND + d];
    out[b * 2 + cc] = s;
  }
}

// ---------------------------------------------------------------------------
extern "C" void kernel_launch(void* const* d_in, const int* in_sizes, int n_in,
                              void* d_out, int out_size, void* d_ws, size_t ws_size,
                              hipStream_t stream) {
  (void)in_sizes; (void)n_in; (void)out_size; (void)ws_size;
  const int*   diag   = (const int*)d_in[0];
  const int*   drug   = (const int*)d_in[1];
  const int*   lab    = (const int*)d_in[2];
  const int*   proc   = (const int*)d_in[3];
  const float* demo   = (const float*)d_in[4];
  const float* emb1   = (const float*)d_in[5];
  const float* emb2   = (const float*)d_in[6];
  const float* emb3   = (const float*)d_in[7];
  const float* emb4   = (const float*)d_in[8];
  const float* demo_W = (const float*)d_in[9];
  const float* demo_b = (const float*)d_in[10];
  const float* wx     = (const float*)d_in[11];
  const float* Wh     = (const float*)d_in[12];
  const float* bh     = (const float*)d_in[13];
  const float* W_ih   = (const float*)d_in[14];
  const float* W_hh   = (const float*)d_in[15];
  const float* b_ih   = (const float*)d_in[16];
  const float* b_hh   = (const float*)d_in[17];
  const float* out_W  = (const float*)d_in[18];
  const float* out_b  = (const float*)d_in[19];
  float* out = (float*)d_out;

  char* ws = (char*)d_ws;
  size_t off = 0;
  auto alloc = [&](size_t bytes) -> char* {
    char* p = ws + off;
    off = (off + bytes + 255) & ~(size_t)255;
    return p;
  };
  float* x      = (float*)alloc((size_t)NB * NT * ND * 4);   // 12.6 MB
  float* xw     = (float*)alloc((size_t)NB * NT * 4);
  float* x2c    = (float*)alloc((size_t)NB * NT * 4);
  float* csum   = (float*)alloc((size_t)NB * NT * 4);
  float* demo_e = (float*)alloc((size_t)NB * ND * 4);
  unsigned short* pWih = (unsigned short*)alloc((size_t)3 * ND * ND * 2);
  unsigned short* pWhh = (unsigned short*)alloc((size_t)3 * ND * ND * 2);
  unsigned short* pWh  = (unsigned short*)alloc((size_t)NT * ND * 2);

  embed_kernel<<<NB * NT, 256, 0, stream>>>(diag, drug, lab, proc,
                                            emb1, emb2, emb3, emb4, wx, x, xw, x2c);
  csum_kernel<<<1, 256, 0, stream>>>(x2c, csum);
  demo_kernel<<<(NB * ND + 255) / 256, 256, 0, stream>>>(demo, demo_W, demo_b, demo_e);
  repack_kernel<<<(3 * ND * ND + 255) / 256, 256, 0, stream>>>(W_ih, pWih, 3 * ND, ND);
  repack_kernel<<<(3 * ND * ND + 255) / 256, 256, 0, stream>>>(W_hh, pWhh, 3 * ND, ND);
  repack_kernel<<<(NT * ND + 255) / 256, 256, 0, stream>>>(Wh, pWh, NT, ND);
  raim_main<<<NB / BM, NTHREADS, 0, stream>>>(x, xw, csum, demo_e, pWih, pWhh, pWh,
                                              bh, b_ih, b_hh, out_W, out_b, out);
}
